// EdgeConvBlock_51084341018863
// MI455X (gfx1250) — matrix operand order repaired and checked by reading, not compiled, not run on hardware
//
#include <hip/hip_runtime.h>

typedef __attribute__((ext_vector_type(16))) __bf16 bf16x16;
typedef __attribute__((ext_vector_type(8)))  float  f32x8;

#define BATCH  4
#define CIN    64
#define NPTS   16384
#define KNN    16
#define BNEPS  1e-5f

// workspace layout (bytes)
#define QT_OFF  ((size_t)0)
#define KT_OFF  ((size_t)16*1024*1024)
#define WP_OFF  ((size_t)32*1024*1024)
#define BN_OFF  (WP_OFF + 40960)

union FragU { bf16x16 v; uint4 u[2]; };
union H8    { __bf16 h[8]; unsigned u32[4]; uint4 u; };

// packed 2xbf16 max (VOP3P V_PK_MAX_NUM_BF16) — CDNA5 inline asm
__device__ inline unsigned pk_max_bf16(unsigned a, unsigned b)
{
  unsigned d;
  asm("v_pk_max_num_bf16 %0, %1, %2" : "=v"(d) : "v"(a), "v"(b));
  return d;
}

// ---------------------------------------------------------------------------
// (B,C,N) f32  ->  (B,N,C) f32 tiled transpose so gathered columns are
// contiguous 256B rows (gathers then hit the 192MB L2 with full lines).
// grid: (N/64, B, 2) ; z selects query vs key.
// ---------------------------------------------------------------------------
__global__ __launch_bounds__(256) void transpose_cn(
    const float* __restrict__ q, const float* __restrict__ key,
    float* __restrict__ qT, float* __restrict__ keyT)
{
  __shared__ float tile[64][65];
  const int b  = blockIdx.y;
  const int n0 = blockIdx.x * 64;
  const float* src = (blockIdx.z == 0) ? q  : key;
  float*       dst = (blockIdx.z == 0) ? qT : keyT;
  src += (size_t)b * CIN * NPTS;
  dst += (size_t)b * NPTS * CIN;

  for (int i = threadIdx.x; i < 64 * 64; i += 256) {
    int c = i >> 6, nn = i & 63;
    tile[c][nn] = src[(size_t)c * NPTS + n0 + nn];
  }
  __syncthreads();
  for (int i = threadIdx.x; i < 64 * 64; i += 256) {
    int nn = i >> 6, c = i & 63;
    dst[(size_t)(n0 + nn) * CIN + c] = tile[c][nn];
  }
}

// ---------------------------------------------------------------------------
// Pre-pack conv weights into the per-lane bf16 WMMA A-fragment layout
// (contiguous k = 16*half + slot, 16-bit A 16x32), with the folded BN scale multiplied in:
//   y = relu( (s.W) x + b' ),  s = gamma*rsqrt(var+eps) per output row.
// Element t = ((frag*32 + lane)*16 + slot).
// W0: 8192 elems @0, W1: 4096 @8192, W2: 8192 @12288  (20480 bf16 = 40KB)
// ---------------------------------------------------------------------------
__global__ __launch_bounds__(256) void w_pack(
    const float* __restrict__ W0, const float* __restrict__ g0, const float* __restrict__ va0,
    const float* __restrict__ W1, const float* __restrict__ g1, const float* __restrict__ va1,
    const float* __restrict__ W2, const float* __restrict__ g2, const float* __restrict__ va2,
    __bf16* __restrict__ wp)
{
  int t = blockIdx.x * 256 + threadIdx.x;
  if (t >= 20480) return;
  const float *W, *g, *va; int Cin, tK, tl;
  if (t < 8192)       { W = W0; g = g0; va = va0; Cin = 128; tK = 4; tl = t; }
  else if (t < 12288) { W = W1; g = g1; va = va1; Cin = 64;  tK = 2; tl = t - 8192; }
  else                { W = W2; g = g2; va = va2; Cin = 64;  tK = 2; tl = t - 12288; }
  const int s    = tl & 15;
  const int l    = (tl >> 4) & 31;
  const int frag = tl >> 9;
  const int m    = frag / tK;
  const int j    = frag - m * tK;
  const int half = l >> 4, lm = l & 15;
  const int r = s >> 1, pos = s & 1;
  const int klocal = 2 * r + pos + half * 16;
  const int row = m * 16 + lm;
  const int k   = j * 32 + klocal;
  const float scale = g[row] * rsqrtf(va[row] + BNEPS);
  wp[t] = (__bf16)(W[row * Cin + k] * scale);
}

// ---------------------------------------------------------------------------
// Folded BN bias only (scale lives in the packed weights):
//   b' = beta - mean*scale.  Layout: [L0 b(64) | L1 b(64) | L2 b(128)]
// ---------------------------------------------------------------------------
__global__ void bn_pack(
    const float* g0, const float* be0, const float* mu0, const float* va0,
    const float* g1, const float* be1, const float* mu1, const float* va1,
    const float* g2, const float* be2, const float* mu2, const float* va2,
    float* __restrict__ bn)
{
  const int t = threadIdx.x;
  if (t < 64) {
    float s = g0[t] * rsqrtf(va0[t] + BNEPS);
    bn[t] = be0[t] - mu0[t] * s;
  } else if (t < 128) {
    int c = t - 64;
    float s = g1[c] * rsqrtf(va1[c] + BNEPS);
    bn[64 + c] = be1[c] - mu1[c] * s;
  } else {
    int c = t - 128;
    float s = g2[c] * rsqrtf(va2[c] + BNEPS);
    bn[128 + c] = be2[c] - mu2[c] * s;
  }
}

// load the 8-wide folded bias for this lane's rows as the WMMA C initializer
__device__ inline f32x8 bias_init(const float* bptr)
{
  float4 b0 = *(const float4*)(bptr);
  float4 b1 = *(const float4*)(bptr + 4);
  f32x8 c = {b0.x, b0.y, b0.z, b0.w, b1.x, b1.y, b1.z, b1.w};
  return c;
}

// pack 8 f32 -> 8 bf16 with packed ReLU (clamp against +0 via v_pk_max_num_bf16)
__device__ inline uint4 pack_relu_bf16(const f32x8& c)
{
  H8 pk;
#pragma unroll
  for (int v = 0; v < 8; ++v) pk.h[v] = (__bf16)c[v];   // fuses to v_cvt_pk_bf16_f32
#pragma unroll
  for (int i = 0; i < 4; ++i) pk.u32[i] = pk_max_bf16(pk.u32[i], 0u);
  return pk.u;
}

// ---------------------------------------------------------------------------
// Fused EdgeConv main kernel: one wave32 per point, 40 bf16 WMMAs per point.
// ---------------------------------------------------------------------------
__global__ __launch_bounds__(256) void edgeconv_main(
    const float* __restrict__ qT, const float* __restrict__ keyT,
    const int* __restrict__ kind, const __bf16* __restrict__ wp,
    const float* __restrict__ bnp, float* __restrict__ out)
{
  __shared__ __bf16 sW[20480];        // packed (BN-scaled) A-fragments, 40KB
  __shared__ float  sBN[256];         // folded BN bias
  __shared__ __bf16 sAct[8][2048];    // per-wave bounce: acts (16x64) / L2 out (16x128)

  for (int i = threadIdx.x; i < 20480 / 8; i += 256)
    ((uint4*)sW)[i] = ((const uint4*)wp)[i];
  for (int i = threadIdx.x; i < 256 / 4; i += 256)
    ((float4*)sBN)[i] = ((const float4*)bnp)[i];
  __syncthreads();

  const int lane = threadIdx.x & 31;
  const int wave = threadIdx.x >> 5;
  const int half = lane >> 4;          // B-fragment: which 16-channel half
  const int lm   = lane & 15;          // B-fragment: neighbor column
  __bf16* act = sAct[wave];

  for (int p = blockIdx.x * 8 + wave; p < BATCH * NPTS; p += gridDim.x * 8) {
    const int b = p >> 14;
    const int n = p & (NPTS - 1);
    const int nb = kind[p * KNN + lm];                 // neighbor for my column
    const float* qcol = qT  + (size_t)p * CIN;
    const float* kcol = keyT + ((size_t)b * NPTS + nb) * CIN;

    // ---- build layer-0 B operand (edge feature, 128 x 16) as 4 bf16 frags
    bf16x16 bfrag[4];
#pragma unroll
    for (int j = 0; j < 2; ++j) {
      const int cb = j * 32 + half * 16;
      float4 q4[4], k4[4];
#pragma unroll
      for (int v = 0; v < 4; ++v) {
        q4[v] = *(const float4*)(qcol + cb + v * 4);
        k4[v] = *(const float4*)(kcol + cb + v * 4);
      }
      const float* qf = (const float*)q4;
      const float* kf = (const float*)k4;
      FragU d, qq;
#pragma unroll
      for (int s = 0; s < 16; ++s) {
        d.v[s]  = (__bf16)(kf[s] - qf[s]);   // channels 0..63   : key - q
        qq.v[s] = (__bf16)qf[s];             // channels 64..127 : q
      }
      bfrag[j]     = d.v;
      bfrag[2 + j] = qq.v;
    }

    // ---- layer 0: W0(64x128) @ edge(128x16), 16 WMMAs, C preloaded with bias
#pragma unroll
    for (int m = 0; m < 4; ++m) {
      f32x8 c = bias_init(sBN + 0 + m * 16 + half * 8);
#pragma unroll
      for (int j = 0; j < 4; ++j) {
        FragU a;
        const uint4* ap = (const uint4*)(sW + ((m * 4 + j) * 32 + lane) * 16);
        a.u[0] = ap[0]; a.u[1] = ap[1];
        c = __builtin_amdgcn_wmma_f32_16x16x32_bf16(false, a.v, false, bfrag[j],
                                                    (short)0, c, false, false);
      }
      *(uint4*)(act + lm * 64 + m * 16 + half * 8) = pack_relu_bf16(c);
    }
    asm volatile("s_wait_dscnt 0x0" ::: "memory");

    // ---- layer 1: W1(64x64) @ act(64x16), 8 WMMAs
    bf16x16 b1[2];
#pragma unroll
    for (int j = 0; j < 2; ++j) {
      FragU r;
      const uint4* pp = (const uint4*)(act + lm * 64 + j * 32 + half * 16);
      r.u[0] = pp[0]; r.u[1] = pp[1];
      b1[j] = r.v;
    }
#pragma unroll
    for (int m = 0; m < 4; ++m) {
      f32x8 c = bias_init(sBN + 64 + m * 16 + half * 8);
#pragma unroll
      for (int j = 0; j < 2; ++j) {
        FragU a;
        const uint4* ap = (const uint4*)(sW + 8192 + ((m * 2 + j) * 32 + lane) * 16);
        a.u[0] = ap[0]; a.u[1] = ap[1];
        c = __builtin_amdgcn_wmma_f32_16x16x32_bf16(false, a.v, false, b1[j],
                                                    (short)0, c, false, false);
      }
      *(uint4*)(act + lm * 64 + m * 16 + half * 8) = pack_relu_bf16(c);
    }
    asm volatile("s_wait_dscnt 0x0" ::: "memory");

    // ---- layer 2: W2(128x64) @ act(64x16), 16 WMMAs
    bf16x16 b2[2];
#pragma unroll
    for (int j = 0; j < 2; ++j) {
      FragU r;
      const uint4* pp = (const uint4*)(act + lm * 64 + j * 32 + half * 16);
      r.u[0] = pp[0]; r.u[1] = pp[1];
      b2[j] = r.v;
    }
#pragma unroll
    for (int m = 0; m < 8; ++m) {
      f32x8 c = bias_init(sBN + 128 + m * 16 + half * 8);
#pragma unroll
      for (int j = 0; j < 2; ++j) {
        FragU a;
        const uint4* ap = (const uint4*)(sW + 12288 + ((m * 2 + j) * 32 + lane) * 16);
        a.u[0] = ap[0]; a.u[1] = ap[1];
        c = __builtin_amdgcn_wmma_f32_16x16x32_bf16(false, a.v, false, b2[j],
                                                    (short)0, c, false, false);
      }
      // raw bf16 (no ReLU yet: relu(max) == max(relu)) to [col][ch] layout
      H8 pk;
#pragma unroll
      for (int v = 0; v < 8; ++v) pk.h[v] = (__bf16)c[v];
      *(uint4*)(act + lm * 128 + m * 16 + half * 8) = pk.u;
    }
    asm volatile("s_wait_dscnt 0x0" ::: "memory");

    // ---- max over the 16 neighbor columns, packed bf16, 4 channels per lane
    {
      const int ch0 = lane * 4;
      uint2 acc2 = *(const uint2*)(act + 0 * 128 + ch0);
#pragma unroll
      for (int col = 1; col < 16; ++col) {
        uint2 w = *(const uint2*)(act + col * 128 + ch0);
        acc2.x = pk_max_bf16(acc2.x, w.x);
        acc2.y = pk_max_bf16(acc2.y, w.y);
      }
      // exact bf16 -> f32 (shift into high bits), then final ReLU
      float f0 = __uint_as_float(acc2.x << 16);
      float f1 = __uint_as_float(acc2.x & 0xffff0000u);
      float f2 = __uint_as_float(acc2.y << 16);
      float f3 = __uint_as_float(acc2.y & 0xffff0000u);
      float* op = out + ((size_t)b * 128 + ch0) * NPTS + n;
      op[0 * NPTS] = fmaxf(f0, 0.f);
      op[1 * NPTS] = fmaxf(f1, 0.f);
      op[2 * NPTS] = fmaxf(f2, 0.f);
      op[3 * NPTS] = fmaxf(f3, 0.f);
    }
  }
}

// ---------------------------------------------------------------------------
extern "C" void kernel_launch(void* const* d_in, const int* in_sizes, int n_in,
                              void* d_out, int out_size, void* d_ws, size_t ws_size,
                              hipStream_t stream)
{
  const float* q   = (const float*)d_in[0];
  const float* key = (const float*)d_in[1];
  const int*   ki  = (const int*)d_in[2];
  const float* W0  = (const float*)d_in[3];
  const float* g0  = (const float*)d_in[4];
  const float* be0 = (const float*)d_in[5];
  const float* mu0 = (const float*)d_in[6];
  const float* va0 = (const float*)d_in[7];
  const float* W1  = (const float*)d_in[8];
  const float* g1  = (const float*)d_in[9];
  const float* be1 = (const float*)d_in[10];
  const float* mu1 = (const float*)d_in[11];
  const float* va1 = (const float*)d_in[12];
  const float* W2  = (const float*)d_in[13];
  const float* g2  = (const float*)d_in[14];
  const float* be2 = (const float*)d_in[15];
  const float* mu2 = (const float*)d_in[16];
  const float* va2 = (const float*)d_in[17];

  char* ws = (char*)d_ws;
  float*  qT   = (float*)(ws + QT_OFF);
  float*  keyT = (float*)(ws + KT_OFF);
  __bf16* wpk  = (__bf16*)(ws + WP_OFF);
  float*  bnp  = (float*)(ws + BN_OFF);
  float*  out  = (float*)d_out;

  transpose_cn<<<dim3(NPTS / 64, BATCH, 2), 256, 0, stream>>>(q, key, qT, keyT);
  w_pack<<<dim3((20480 + 255) / 256), 256, 0, stream>>>(W0, g0, va0,
                                                        W1, g1, va1,
                                                        W2, g2, va2, wpk);
  bn_pack<<<dim3(1), 256, 0, stream>>>(g0, be0, mu0, va0,
                                       g1, be1, mu1, va1,
                                       g2, be2, mu2, va2, bnp);
  edgeconv_main<<<dim3(2048), 256, 0, stream>>>(qT, keyT, ki, wpk, bnp, out);
}
